// TripletLoss_43336220017242
// MI455X (gfx1250) — matrix operand order repaired and checked by reading, not compiled, not run on hardware
//
#include <hip/hip_runtime.h>
#include <hip/hip_bf16.h>
#include <math.h>

typedef __attribute__((ext_vector_type(16))) __bf16 v16bf;
typedef __attribute__((ext_vector_type(8)))  __bf16 v8bf;
typedef __attribute__((ext_vector_type(8)))  float  v8f;

#define N 8192
#define D 128
#define NT (N / 16)      // 512 column tiles
#define MARGIN 0.3f

// ---------------------------------------------------------------------------
// Kernel 1: split fp32 inputs into hi/lo bf16 planes, compute row sq-norms.
// grid = N blocks, 128 threads (one per feature).
// ---------------------------------------------------------------------------
__global__ void prep_kernel(const float* __restrict__ x,
                            __bf16* __restrict__ hi,
                            __bf16* __restrict__ lo,
                            float* __restrict__ sq) {
    const int row = blockIdx.x;
    const int k   = threadIdx.x;
    const float v = x[row * D + k];
    const __bf16 h = (__bf16)v;
    const __bf16 l = (__bf16)(v - (float)h);
    hi[row * D + k] = h;
    lo[row * D + k] = l;

    __shared__ float red[D];
    red[k] = v * v;
    __syncthreads();
    #pragma unroll
    for (int s = D / 2; s > 0; s >>= 1) {
        if (k < s) red[k] += red[k + s];
        __syncthreads();
    }
    if (k == 0) sq[row] = red[0];
}

__global__ void zero_kernel(float* out) {
    if (threadIdx.x == 0) out[0] = 0.0f;
}

// ---------------------------------------------------------------------------
// Kernel 2: fused Gram + masked row reductions + loss.
// grid = 512 blocks (16 rows each), 256 threads = 8 waves.
// Each wave sweeps column tiles ct = w, w+8, ..., computing a 16x16 Gram tile
// with 12 bf16 WMMAs (hi*hi + hi*lo + lo*hi over 4 K-chunks of 32).
// Accumulation epilogue is fully branchless (cndmask/min/max) so it can
// co-execute with the WMMA (TRANS) pipe.
// ---------------------------------------------------------------------------
__global__ void __launch_bounds__(256)
triplet_kernel(const __bf16* __restrict__ hi,
               const __bf16* __restrict__ lo,
               const float* __restrict__ sq,
               const int* __restrict__ tgt,
               float* __restrict__ out) {
    const int r0   = blockIdx.x * 16;
    const int lane = threadIdx.x & 31;
    const int w    = threadIdx.x >> 5;

    // ---- A fragments for this block's 16 rows (contiguous K order, matches B):
    // lane l: M = l&15 ; VGPR0-3 hold K = koff..koff+7, VGPR4-7 hold K = koff+8..koff+15
    // where koff = (l>>4)*16.  All contiguous bf16 runs of 8.
    const int m    = lane & 15;
    const int koff = (lane >> 4) * 16;
    const int arow = r0 + m;
    v16bf ah[4], al[4];
    #pragma unroll
    for (int kc = 0; kc < 4; ++kc) {
        const v8bf h0 = *(const v8bf*)(hi + (size_t)arow * D + kc * 32 + koff);
        const v8bf h1 = *(const v8bf*)(hi + (size_t)arow * D + kc * 32 + 8 + koff);
        const v8bf l0 = *(const v8bf*)(lo + (size_t)arow * D + kc * 32 + koff);
        const v8bf l1 = *(const v8bf*)(lo + (size_t)arow * D + kc * 32 + 8 + koff);
        #pragma unroll
        for (int t = 0; t < 8; ++t) {
            ah[kc][t] = h0[t]; ah[kc][8 + t] = h1[t];
            al[kc][t] = l0[t]; al[kc][8 + t] = l1[t];
        }
    }

    // ---- Per-lane row metadata: C/D layout puts M = rbase+v in this lane.
    const int rbase = (lane >> 4) * 8;
    float srow[8]; int trow[8];
    #pragma unroll
    for (int v = 0; v < 8; ++v) {
        srow[v] = sq[r0 + rbase + v];
        trow[v] = tgt[r0 + rbase + v];
    }

    // ---- Accumulators (per lane = per output column, 8 rows each)
    float sum_p[8], max_p[8], sum_n[8], min_n[8], cnt_p[8];
    #pragma unroll
    for (int v = 0; v < 8; ++v) {
        sum_p[v] = 0.0f; max_p[v] = -INFINITY;
        sum_n[v] = 0.0f; min_n[v] =  INFINITY;
        cnt_p[v] = 0.0f;
    }

    // ---- Column-tile sweep (uniform per wave -> EXEC all ones at WMMA)
    const int n  = lane & 15;
    const int kb = (lane >> 4) * 16;   // B 32x16 layout: lane holds K=kb..kb+15 of col n
    for (int ct = w; ct < NT; ct += 8) {
        const int col = ct * 16 + n;
        v16bf bh[4], bl[4];
        #pragma unroll
        for (int kc = 0; kc < 4; ++kc) {
            bh[kc] = *(const v16bf*)(hi + (size_t)col * D + kc * 32 + kb);
            bl[kc] = *(const v16bf*)(lo + (size_t)col * D + kc * 32 + kb);
        }

        v8f c = {};
        #pragma unroll
        for (int kc = 0; kc < 4; ++kc) {
            c = __builtin_amdgcn_wmma_f32_16x16x32_bf16(false, ah[kc], false, bh[kc],
                                                        (short)0, c, false, false);
            c = __builtin_amdgcn_wmma_f32_16x16x32_bf16(false, ah[kc], false, bl[kc],
                                                        (short)0, c, false, false);
            c = __builtin_amdgcn_wmma_f32_16x16x32_bf16(false, al[kc], false, bh[kc],
                                                        (short)0, c, false, false);
        }

        const float sqj = sq[col];
        const int   tj  = tgt[col];
        #pragma unroll
        for (int v = 0; v < 8; ++v) {
            const int i = r0 + rbase + v;
            // dsq = dist^2 = clip(d2, 1e-12)  (no sqrt needed anywhere)
            const float dsq = fmaxf(srow[v] + sqj - 2.0f * c[v], 1e-12f);
            const bool same    = (trow[v] == tj);
            const bool valid_p = same & (i != col);   // drop self-distance
            // fully branchless: v_cndmask feeding add/max/min
            sum_p[v] += valid_p ? dsq  : 0.0f;
            cnt_p[v] += valid_p ? 1.0f : 0.0f;
            max_p[v]  = fmaxf(max_p[v], valid_p ? dsq :  -INFINITY);
            sum_n[v] += same    ? 0.0f : dsq;
            min_n[v]  = fminf(min_n[v], same    ? INFINITY : dsq);
        }
    }

    // ---- Reduce across the 16 lanes sharing each row group
    #pragma unroll
    for (int off = 8; off > 0; off >>= 1) {
        #pragma unroll
        for (int v = 0; v < 8; ++v) {
            sum_p[v] += __shfl_xor(sum_p[v], off, 16);
            max_p[v]  = fmaxf(max_p[v], __shfl_xor(max_p[v], off, 16));
            sum_n[v] += __shfl_xor(sum_n[v], off, 16);
            min_n[v]  = fminf(min_n[v], __shfl_xor(min_n[v], off, 16));
            cnt_p[v] += __shfl_xor(cnt_p[v], off, 16);
        }
    }

    // ---- Cross-wave combine via LDS
    __shared__ float s_sp[16][8], s_mp[16][8], s_sn[16][8], s_mn[16][8], s_cp[16][8];
    if ((lane & 15) == 0) {
        #pragma unroll
        for (int v = 0; v < 8; ++v) {
            s_sp[rbase + v][w] = sum_p[v];
            s_mp[rbase + v][w] = max_p[v];
            s_sn[rbase + v][w] = sum_n[v];
            s_mn[rbase + v][w] = min_n[v];
            s_cp[rbase + v][w] = cnt_p[v];
        }
    }
    __syncthreads();

    if (threadIdx.x < 16) {
        const int r = threadIdx.x;
        float sp = 0.0f, mp = -INFINITY, sn = 0.0f, mn = INFINITY, cp = 0.0f;
        #pragma unroll
        for (int w2 = 0; w2 < 8; ++w2) {
            sp += s_sp[r][w2];
            mp  = fmaxf(mp, s_mp[r][w2]);
            sn += s_sn[r][w2];
            mn  = fminf(mn, s_mn[r][w2]);
            cp += s_cp[r][w2];
        }
        const float sigma_p = sp / cp;
        const float dist_ap = mp / sigma_p + 0.5f * __logf(sigma_p);
        const float cn      = (float)(N - 1) - cp;
        const float sigma_n = sn / cn;
        const float dist_an = mn / sigma_n + 0.5f * __logf(sigma_n);
        float li = fmaxf(dist_ap - dist_an + MARGIN, 0.0f);
        #pragma unroll
        for (int off = 8; off > 0; off >>= 1)
            li += __shfl_xor(li, off, 16);
        if (r == 0)
            atomicAdd(out, li * (1.0f / (float)N));
    }
}

// ---------------------------------------------------------------------------
extern "C" void kernel_launch(void* const* d_in, const int* in_sizes, int n_in,
                              void* d_out, int out_size, void* d_ws, size_t ws_size,
                              hipStream_t stream) {
    const float* x   = (const float*)d_in[0];
    const int*   tgt = (const int*)d_in[1];
    float*       out = (float*)d_out;

    char* ws = (char*)d_ws;
    __bf16* hi = (__bf16*)ws;                                         // 2 MB
    __bf16* lo = (__bf16*)(ws + (size_t)N * D * sizeof(__bf16));      // 2 MB
    float*  sq = (float*) (ws + (size_t)2 * N * D * sizeof(__bf16));  // 32 KB

    prep_kernel<<<N, D, 0, stream>>>(x, hi, lo, sq);
    zero_kernel<<<1, 1, 0, stream>>>(out);
    triplet_kernel<<<N / 16, 256, 0, stream>>>(hi, lo, sq, tgt, out);
}
